// MixtureOfExperts_31078383354140
// MI455X (gfx1250) — matrix operand order repaired and checked by reading, not compiled, not run on hardware
//
#include <hip/hip_runtime.h>
#include <hip/hip_bf16.h>
#include <cstdint>
#include <cstddef>

// ---------------------------------------------------------------------------
// MoE (B=8192, D=1024, E=12, H=512, O=256), eval mode.
// bf16 WMMA (16x16x32, f32 accum) for every GEMM; weights pre-transposed to
// [N][K] during the f32->bf16 cast; async double-buffered LDS staging via
// GLOBAL_LOAD_ASYNC_TO_LDS_B128 (ASYNCcnt); fused bias+ReLU+BN epilogue
// (mode 0) or gate-weighted accumulate into the output (mode 1).
// GEMM: 128 threads (4 waves, 2x2), wave tile 64x64 (4x4 WMMA sub-tiles)
// -> 16 v_wmma per 16 ds_load_b128 per K-step, 4x reuse of A and B fragments.
// ---------------------------------------------------------------------------

#define MOE_EPS 1e-5f

typedef __attribute__((ext_vector_type(16))) __bf16 v16bf;
typedef __attribute__((ext_vector_type(8)))  float  v8f;

union FragAB {
    v16bf v;
    uint4 u[2];
};

// ---------------------------------------------------------------- utilities
__global__ void cast_f32_to_bf16(const float* __restrict__ in,
                                 __bf16* __restrict__ out, int n) {
    for (int i = blockIdx.x * blockDim.x + threadIdx.x; i < n;
         i += gridDim.x * blockDim.x)
        out[i] = (__bf16)in[i];
}

__global__ void zero_f32(float* __restrict__ p, int n) {
    for (int i = blockIdx.x * blockDim.x + threadIdx.x; i < n;
         i += gridDim.x * blockDim.x)
        p[i] = 0.0f;
}

// cast + transpose: in = f32 [Kd][Nd] row-major, out = bf16 [Nd][Kd] row-major
// grid = (Kd/32, Nd/32, nmat), block = (32, 8)
__global__ void cast_transpose_f32_to_bf16(const float* __restrict__ in,
                                           __bf16* __restrict__ out,
                                           int Kd, int Nd) {
    __shared__ float tile[32][33];
    const size_t mat = (size_t)blockIdx.z * Kd * Nd;
    in  += mat;
    out += mat;
    const int k0 = blockIdx.x * 32;
    const int n0 = blockIdx.y * 32;
    const int tx = threadIdx.x;   // 0..31
    const int ty = threadIdx.y;   // 0..7
#pragma unroll
    for (int i = 0; i < 32; i += 8)
        tile[ty + i][tx] = in[(size_t)(k0 + ty + i) * Nd + n0 + tx];
    __syncthreads();
#pragma unroll
    for (int i = 0; i < 32; i += 8)
        out[(size_t)(n0 + ty + i) * Kd + k0 + tx] = (__bf16)tile[tx][ty + i];
}

// ------------------------------------------------------- main WMMA GEMM
// C[M,N] = A[M,K] (bf16 row-major) * Wt[N,K] (bf16 row-major, pre-transposed)
// mode 0: y = BN(relu(acc + bias)) -> out_bf (bf16)
// mode 1: out_f32[b,n] += gates[b*12+gate_e] * (acc + bias)
// Block: 128 threads (4 waves, 2x2). Block tile 128x128, wave tile 64x64.
__global__ __launch_bounds__(128)
void gemm_bf16_wmma(const __bf16* __restrict__ A, const __bf16* __restrict__ Wt,
                    const float* __restrict__ bias,
                    const float* __restrict__ bn_g, const float* __restrict__ bn_b,
                    const float* __restrict__ bn_m, const float* __restrict__ bn_v,
                    const float* __restrict__ gates, int gate_e,
                    __bf16* __restrict__ out_bf, float* __restrict__ out_f32,
                    int M, int N, int K, int mode)
{
    // Double-buffered LDS tiles, rows padded to 40 bf16 (80B):
    // conflict-free b128 fragment loads, 16B-aligned async targets.
    __shared__ __bf16 sA[2][128 * 40];   // [buf][row(M) 0..127][K 0..31]
    __shared__ __bf16 sB[2][128 * 40];   // [buf][row(N) 0..127][K 0..31]

    const int t    = threadIdx.x;        // 0..127
    const int lane = t & 31;
    const int wave = t >> 5;             // 0..3
    const int kg   = lane >> 4;          // K-half selector
    const int ln   = lane & 15;
    const int wm0  = (wave >> 1) * 64;   // wave row origin in block tile
    const int wn0  = (wave & 1) * 64;    // wave col origin in block tile
    const int bm   = blockIdx.y * 128;
    const int bn   = blockIdx.x * 128;

    // ---- async staging: each tile = 512 16B segments (128 rows x 4 segs);
    // 128 threads own 4 segments of each tile: rows m1+32j, j=0..3.
    const int m1 = t >> 2;               // 0..31
    const int s1 = t & 3;                // 0..3
    const __bf16* gA[4];
    const __bf16* gB[4];
#pragma unroll
    for (int j = 0; j < 4; ++j) {
        gA[j] = A  + (size_t)(bm + m1 + 32 * j) * K + 8 * s1;
        gB[j] = Wt + (size_t)(bn + m1 + 32 * j) * K + 8 * s1;
    }
    const uint32_t BUFB = 128u * 40u * 2u;                 // 10240 B per buffer
    const uint32_t segO = (uint32_t)(m1 * 40 + 8 * s1) * 2u;
    const uint32_t lA0  = (uint32_t)(uintptr_t)(&sA[0][0]) + segO;
    const uint32_t lB0  = (uint32_t)(uintptr_t)(&sB[0][0]) + segO;
    const uint32_t rowJ = 32u * 40u * 2u;                  // 2560 B per 32 rows

    auto issue_tile = [&](uint32_t boff) {
#pragma unroll
        for (int j = 0; j < 4; ++j)
            asm volatile("global_load_async_to_lds_b128 %0, %1, off"
                         :: "v"(lA0 + boff + j * rowJ), "v"(gA[j]) : "memory");
#pragma unroll
        for (int j = 0; j < 4; ++j)
            asm volatile("global_load_async_to_lds_b128 %0, %1, off"
                         :: "v"(lB0 + boff + j * rowJ), "v"(gB[j]) : "memory");
    };

    v8f acc[4][4];
#pragma unroll
    for (int i = 0; i < 4; ++i)
#pragma unroll
        for (int j = 0; j < 4; ++j)
#pragma unroll
            for (int r = 0; r < 8; ++r) acc[i][j][r] = 0.0f;

    // prefetch tile 0 into buffer 0
    issue_tile(0);
#pragma unroll
    for (int j = 0; j < 4; ++j) { gA[j] += 32; gB[j] += 32; }

    const int nk = K >> 5;
    for (int kt = 0; kt < nk; ++kt) {
        // own async copies done, then block-wide visibility
        asm volatile("s_wait_asynccnt 0x0" ::: "memory");
        __syncthreads();

        // start next tile into the other buffer (overlaps with compute below)
        if (kt + 1 < nk) {
            issue_tile(((kt + 1) & 1) ? BUFB : 0u);
#pragma unroll
            for (int j = 0; j < 4; ++j) { gA[j] += 32; gB[j] += 32; }
        }

        const int p = kt & 1;

        // ---- fragment loads (2x ds_load_b128 each, ISA 16-bit A/B layouts)
        FragAB fa[4], fb[4];
#pragma unroll
        for (int ti = 0; ti < 4; ++ti) {
            const __bf16* pa = &sA[p][(wm0 + 16 * ti + ln) * 40 + 8 * kg];
            fa[ti].u[0] = *reinterpret_cast<const uint4*>(pa);       // K = 8kg..
            fa[ti].u[1] = *reinterpret_cast<const uint4*>(pa + 16);  // K = 16+8kg..
        }
#pragma unroll
        for (int tj = 0; tj < 4; ++tj) {
            const __bf16* pb = &sB[p][(wn0 + 16 * tj + ln) * 40 + 8 * kg];
            fb[tj].u[0] = *reinterpret_cast<const uint4*>(pb);       // K = 8kg..
            fb[tj].u[1] = *reinterpret_cast<const uint4*>(pb + 16);
        }

        // ---- 16 WMMAs per wave per k-step (1.0 ds_load per wmma)
#pragma unroll
        for (int ti = 0; ti < 4; ++ti)
#pragma unroll
            for (int tj = 0; tj < 4; ++tj)
                acc[ti][tj] = __builtin_amdgcn_wmma_f32_16x16x32_bf16(
                    false, fa[ti].v, false, fb[tj].v,
                    (short)0, acc[ti][tj], false, false);
    }

    // ---- epilogue (C/D layout: VGPR r, lanes 0-15 -> M=r, lanes 16-31 -> M=8+r)
#pragma unroll
    for (int ti = 0; ti < 4; ++ti) {
#pragma unroll
        for (int tj = 0; tj < 4; ++tj) {
            int col  = bn + wn0 + 16 * tj + ln;
            float bi = bias[col];
            if (mode == 0) {
                float inv = rsqrtf(bn_v[col] + MOE_EPS) * bn_g[col];
                float bb  = bn_b[col];
                float mm  = bn_m[col];
#pragma unroll
                for (int r = 0; r < 8; ++r) {
                    int row = bm + wm0 + 16 * ti + 8 * kg + r;
                    float y = acc[ti][tj][r] + bi;
                    y = y > 0.0f ? y : 0.0f;
                    y = (y - mm) * inv + bb;
                    out_bf[(size_t)row * N + col] = (__bf16)y;
                }
            } else {
#pragma unroll
                for (int r = 0; r < 8; ++r) {
                    int row = bm + wm0 + 16 * ti + 8 * kg + r;
                    float y = acc[ti][tj][r] + bi;
                    out_f32[(size_t)row * N + col] +=
                        gates[(size_t)row * 12 + gate_e] * y;
                }
            }
        }
    }
}

// ------------------------------------------- gate head: Linear(128,12)+softmax
__global__ void gate3_softmax(const __bf16* __restrict__ h2,
                              const float* __restrict__ gw3,
                              const float* __restrict__ gb3,
                              float* __restrict__ gates, int Bn)
{
    int b = blockIdx.x * blockDim.x + threadIdx.x;
    if (b >= Bn) return;
    float lg[12];
#pragma unroll
    for (int e = 0; e < 12; ++e) lg[e] = gb3[e];
    for (int k = 0; k < 128; ++k) {
        float hv = (float)h2[(size_t)b * 128 + k];
#pragma unroll
        for (int e = 0; e < 12; ++e) lg[e] += hv * gw3[k * 12 + e];
    }
    float mx = lg[0];
#pragma unroll
    for (int e = 1; e < 12; ++e) mx = fmaxf(mx, lg[e]);
    float s = 0.0f;
#pragma unroll
    for (int e = 0; e < 12; ++e) { lg[e] = __expf(lg[e] - mx); s += lg[e]; }
    float inv = 1.0f / s;
#pragma unroll
    for (int e = 0; e < 12; ++e) gates[(size_t)b * 12 + e] = lg[e] * inv;
}

// ---------------------------------------------------------------------------
extern "C" void kernel_launch(void* const* d_in, const int* in_sizes, int n_in,
                              void* d_out, int out_size, void* d_ws, size_t ws_size,
                              hipStream_t stream)
{
    (void)in_sizes; (void)n_in; (void)out_size;
    const int Bn = 8192, D = 1024, E = 12, H = 512, O = 256;

    const float* x      = (const float*)d_in[0];
    const float* gw1    = (const float*)d_in[1];
    const float* gb1    = (const float*)d_in[2];
    const float* gbn1_g = (const float*)d_in[3];
    const float* gbn1_b = (const float*)d_in[4];
    const float* gbn1_m = (const float*)d_in[5];
    const float* gbn1_v = (const float*)d_in[6];
    const float* gw2    = (const float*)d_in[7];
    const float* gb2    = (const float*)d_in[8];
    const float* gbn2_g = (const float*)d_in[9];
    const float* gbn2_b = (const float*)d_in[10];
    const float* gbn2_m = (const float*)d_in[11];
    const float* gbn2_v = (const float*)d_in[12];
    const float* gw3    = (const float*)d_in[13];
    const float* gb3    = (const float*)d_in[14];
    const float* ew1    = (const float*)d_in[15];
    const float* eb1    = (const float*)d_in[16];
    const float* ebn1_g = (const float*)d_in[17];
    const float* ebn1_b = (const float*)d_in[18];
    const float* ebn1_m = (const float*)d_in[19];
    const float* ebn1_v = (const float*)d_in[20];
    const float* ew2    = (const float*)d_in[21];
    const float* eb2    = (const float*)d_in[22];
    const float* ebn2_g = (const float*)d_in[23];
    const float* ebn2_b = (const float*)d_in[24];
    const float* ebn2_m = (const float*)d_in[25];
    const float* ebn2_v = (const float*)d_in[26];
    const float* ew3    = (const float*)d_in[27];
    const float* eb3    = (const float*)d_in[28];
    const float* ebn3_g = (const float*)d_in[29];
    const float* ebn3_b = (const float*)d_in[30];
    const float* ebn3_m = (const float*)d_in[31];
    const float* ebn3_v = (const float*)d_in[32];
    const float* ew4    = (const float*)d_in[33];
    const float* eb4    = (const float*)d_in[34];
    float* out = (float*)d_out;

    // ---- workspace carve-out (256B aligned sub-buffers)
    char*  ws  = (char*)d_ws;
    size_t off = 0;
    auto take = [&](size_t bytes) -> char* {
        char* p = ws + off;
        off += (bytes + 255) & ~(size_t)255;
        return p;
    };
    __bf16* x_bf   = (__bf16*)take((size_t)Bn * D * 2);
    __bf16* gw1t   = (__bf16*)take((size_t)D * 256 * 2);          // [256][D]
    __bf16* gw2t   = (__bf16*)take((size_t)256 * 128 * 2);        // [128][256]
    __bf16* ew1t   = (__bf16*)take((size_t)E * D * H * 2);        // [e][H][D]
    __bf16* ew2t   = (__bf16*)take((size_t)E * H * H * 2);        // [e][H][H]
    __bf16* ew3t   = (__bf16*)take((size_t)E * H * (H / 2) * 2);  // [e][H/2][H]
    __bf16* ew4t   = (__bf16*)take((size_t)E * (H / 2) * O * 2);  // [e][O][H/2]
    __bf16* h1     = (__bf16*)take((size_t)Bn * 256 * 2);
    __bf16* h2     = (__bf16*)take((size_t)Bn * 128 * 2);
    float*  gates  = (float*)take((size_t)Bn * E * 4);
    __bf16* a1     = (__bf16*)take((size_t)Bn * H * 2);
    __bf16* a2     = (__bf16*)take((size_t)Bn * H * 2);
    __bf16* a3     = (__bf16*)take((size_t)Bn * (H / 2) * 2);
    if (off > ws_size) return;   // workspace too small; nothing safe to do

    // activations: plain cast
    {
        int n = Bn * D;
        int blocks = (n + 255) / 256; if (blocks > 4096) blocks = 4096;
        cast_f32_to_bf16<<<blocks, 256, 0, stream>>>(x, x_bf, n);
    }
    // weights: cast + transpose to [N][K] (per matrix, batched over z)
    auto castT = [&](const float* src, __bf16* dst, int Kd, int Nd, int nmat) {
        dim3 grid(Kd / 32, Nd / 32, nmat);
        cast_transpose_f32_to_bf16<<<grid, dim3(32, 8), 0, stream>>>(src, dst, Kd, Nd);
    };
    castT(gw1, gw1t, D, 256, 1);
    castT(gw2, gw2t, 256, 128, 1);
    castT(ew1, ew1t, D, H, E);
    castT(ew2, ew2t, H, H, E);
    castT(ew3, ew3t, H, H / 2, E);
    castT(ew4, ew4t, H / 2, O, E);

    auto gemm = [&](const __bf16* Am, const __bf16* Wm, const float* bias,
                    const float* g, const float* bb, const float* m, const float* v,
                    const float* gt, int ge, __bf16* ob, float* of,
                    int M_, int N_, int K_, int mode) {
        dim3 grid(N_ / 128, M_ / 128);
        gemm_bf16_wmma<<<grid, 128, 0, stream>>>(Am, Wm, bias, g, bb, m, v,
                                                 gt, ge, ob, of, M_, N_, K_, mode);
    };

    // ---- gate network
    gemm(x_bf, gw1t, gb1, gbn1_g, gbn1_b, gbn1_m, gbn1_v,
         nullptr, 0, h1, nullptr, Bn, 256, D, 0);
    gemm(h1, gw2t, gb2, gbn2_g, gbn2_b, gbn2_m, gbn2_v,
         nullptr, 0, h2, nullptr, Bn, 128, 256, 0);
    gate3_softmax<<<Bn / 256, 256, 0, stream>>>(h2, gw3, gb3, gates, Bn);

    // ---- output accumulator
    zero_f32<<<2048, 256, 0, stream>>>(out, Bn * O);

    // ---- experts: 4-layer GEMM chain each, gate-weighted accumulate at the end
    for (int e = 0; e < E; ++e) {
        gemm(x_bf, ew1t + (size_t)e * D * H, eb1 + (size_t)e * H,
             ebn1_g + (size_t)e * H, ebn1_b + (size_t)e * H,
             ebn1_m + (size_t)e * H, ebn1_v + (size_t)e * H,
             nullptr, 0, a1, nullptr, Bn, H, D, 0);
        gemm(a1, ew2t + (size_t)e * H * H, eb2 + (size_t)e * H,
             ebn2_g + (size_t)e * H, ebn2_b + (size_t)e * H,
             ebn2_m + (size_t)e * H, ebn2_v + (size_t)e * H,
             nullptr, 0, a2, nullptr, Bn, H, H, 0);
        gemm(a2, ew3t + (size_t)e * H * (H / 2), eb3 + (size_t)e * (H / 2),
             ebn3_g + (size_t)e * (H / 2), ebn3_b + (size_t)e * (H / 2),
             ebn3_m + (size_t)e * (H / 2), ebn3_v + (size_t)e * (H / 2),
             nullptr, 0, a3, nullptr, Bn, H / 2, H, 0);
        gemm(a3, ew4t + (size_t)e * (H / 2) * O, eb4 + (size_t)e * O,
             nullptr, nullptr, nullptr, nullptr,
             gates, e, nullptr, out, Bn, O, H / 2, 1);
    }
}
